// IGMambaModule_Symmetric_Experts_77300821394074
// MI455X (gfx1250) — matrix operand order repaired and checked by reading, not compiled, not run on hardware
//
#include <hip/hip_runtime.h>
#include <hip/hip_bf16.h>
#include <math.h>

#define BB 2
#define CC 64
#define HH 224
#define WWD 224
#define HP 112
#define WP 112
#define LSEQ (HP*WP)          // 12544
#define DIN 128
#define HWF (HH*WWD)          // 50176
#define HWH (HP*WP)           // 12544

static inline int CDIV(int a, int b){ return (a + b - 1) / b; }

typedef __attribute__((ext_vector_type(16))) _Float16 v16h;
typedef __attribute__((ext_vector_type(8)))  float    v8f;
typedef __attribute__((ext_vector_type(4)))  float    f4v;

#if __has_builtin(__builtin_amdgcn_global_load_async_to_lds_b32) && __has_builtin(__builtin_amdgcn_s_wait_asynccnt)
#define HAVE_ASYNC_LDS 1
#endif

// ---- WMMA fragment k order used here (contiguous, not the ISA table order; wave32) ----
// A (16x32 f16): element p of v16h -> k = p + 16*khalf  (fully contiguous)
// B (32x16 f16): element p -> k = p + 16*khalf  (fully contiguous)

__device__ __forceinline__ float sigmoidf_(float v){ return 1.f/(1.f+expf(-v)); }

// half->full bilinear (half-pixel centers, edge clamp), matches jax resize 2x
__device__ __forceinline__ float bil2x(const float* __restrict__ p, int nh, int nw, int oh, int ow){
  float sh = oh*0.5f - 0.25f, sw = ow*0.5f - 0.25f;
  int h0 = (int)floorf(sh), w0 = (int)floorf(sw);
  float fh = sh - (float)h0, fw = sw - (float)w0;
  int h0c = min(max(h0,0),nh-1), h1c = min(max(h0+1,0),nh-1);
  int w0c = min(max(w0,0),nw-1), w1c = min(max(w0+1,0),nw-1);
  float v00=p[h0c*nw+w0c], v01=p[h0c*nw+w1c], v10=p[h1c*nw+w0c], v11=p[h1c*nw+w1c];
  return (1.f-fh)*((1.f-fw)*v00+fw*v01) + fh*((1.f-fw)*v10+fw*v11);
}

// ---------------- elementwise / reduction kernels ----------------

__global__ void k_pre(const float* __restrict__ x, const float* __restrict__ illum,
                      const float* __restrict__ grad, const float* __restrict__ iw,
                      const float* __restrict__ ib, const float* __restrict__ alpha_p,
                      float* __restrict__ f_low, float* __restrict__ f_high){
  int idx = blockIdx.x*blockDim.x + threadIdx.x;
  if (idx >= BB*HWF) return;
  int b = idx / HWF, pix = idx - b*HWF;
  int h = pix / WWD, w = pix - h*WWD;
  float ci = bil2x(illum + (size_t)b*HWH, HP, WP, h, w);
  float cg = bil2x(grad  + (size_t)b*HWH, HP, WP, h, w);
  float acc = ib[0] + iw[CC]*ci;
  const float* xb = x + (size_t)b*CC*HWF + pix;
  for (int c=0;c<CC;c++) acc += iw[c]*xb[(size_t)c*HWF];
  float m = sigmoidf_(acc);
  float hi = 1.f + alpha_p[0]*cg;
  for (int c=0;c<CC;c++){
    float xv = xb[(size_t)c*HWF];
    f_low [(size_t)(b*CC+c)*HWF + pix] = xv*m;
    f_high[(size_t)(b*CC+c)*HWF + pix] = xv*hi;
  }
}

__global__ void k_avgpool(const float* __restrict__ f_low, float* __restrict__ fd){
  int idx = blockIdx.x*blockDim.x + threadIdx.x;
  if (idx >= BB*CC*HWH) return;
  int bc = idx / HWH, pix = idx - bc*HWH;
  int i = pix / WP, j = pix - i*WP;
  const float* p = f_low + (size_t)bc*HWF;
  float s = p[(2*i)*WWD + 2*j] + p[(2*i)*WWD + 2*j+1]
          + p[(2*i+1)*WWD + 2*j] + p[(2*i+1)*WWD + 2*j+1];
  fd[idx] = 0.25f*s;
}

__global__ void k_resup(const float* __restrict__ f_low, const float* __restrict__ fd,
                        float* __restrict__ t){
  int idx = blockIdx.x*blockDim.x + threadIdx.x;
  if (idx >= BB*CC*HWF) return;
  int bc = idx / HWF, pix = idx - bc*HWF;
  int h = pix / WWD, w = pix - h*WWD;
  t[idx] = f_low[idx] - bil2x(fd + (size_t)bc*HWH, HP, WP, h, w);
}

__global__ void k_dw3(const float* __restrict__ t, const float* __restrict__ dw,
                      const float* __restrict__ db, float* __restrict__ res){
  int idx = blockIdx.x*blockDim.x + threadIdx.x;
  if (idx >= BB*CC*HWF) return;
  int bc = idx / HWF, pix = idx - bc*HWF;
  int c = bc % CC;
  int h = pix / WWD, w = pix - h*WWD;
  const float* p = t + (size_t)bc*HWF;
  float acc = db[c];
  for (int kh=0;kh<3;kh++){
    int ih = h + kh - 1; if (ih < 0 || ih >= HH) continue;
    for (int kw=0;kw<3;kw++){
      int iw2 = w + kw - 1; if (iw2 < 0 || iw2 >= WWD) continue;
      acc += p[ih*WWD + iw2] * dw[c*9 + kh*3 + kw];
    }
  }
  res[idx] = acc;
}

__global__ void k_rmsscale(const float* __restrict__ fd, float* __restrict__ rmss){
  int m = blockIdx.x*blockDim.x + threadIdx.x;
  if (m >= BB*LSEQ) return;
  int b = m / LSEQ, l = m - b*LSEQ;
  float s = 0.f;
  for (int c=0;c<CC;c++){ float v = fd[(size_t)(b*CC+c)*LSEQ + l]; s += v*v; }
  rmss[m] = rsqrtf(s*(1.f/CC) + 1e-5f);
}

// ---------------- WMMA kernels ----------------

// xz = rmsnorm(xs) @ in_w.T ; M=B*L, K=64, N=256. Weights (16-col slice) in LDS f16.
__global__ void __launch_bounds__(128) k_gemm_inproj(
    const float* __restrict__ fd, const float* __restrict__ rmss,
    const float* __restrict__ g, const float* __restrict__ W,
    float* __restrict__ out){
  __shared__ _Float16 Ws[16*72];     // 16 cols x K=64, stride 72 halfs
  int tid = threadIdx.x;
  int colBase = blockIdx.y*16;
  for (int i = tid; i < 16*64; i += 128){
    int nl = i >> 6, k = i & 63;
    Ws[nl*72 + k] = (_Float16)W[(size_t)(colBase+nl)*64 + k];
  }
  __syncthreads();
  int wave = tid >> 5, lane = tid & 31, khalf = lane >> 4;
  int tileM = blockIdx.x*4 + wave;   // M = 25088 -> exact
  int row = tileM*16 + (lane & 15);
  int col = colBase + (lane & 15);
  int b = row / LSEQ, l = row - b*LSEQ;
  float sc = rmss[row];
  v8f c = {};
  for (int kk=0; kk<64; kk+=32){
    v16h a, bf;
#pragma unroll
    for (int p=0;p<8;p++){
      int k0 = kk + khalf*16 + p;
      int k1 = kk + khalf*16 + 8 + p;
      a[p]   = (_Float16)(fd[(size_t)(b*CC+k0)*LSEQ + l] * sc * g[k0]);
      a[p+8] = (_Float16)(fd[(size_t)(b*CC+k1)*LSEQ + l] * sc * g[k1]);
    }
    const _Float16* wr = &Ws[(lane&15)*72 + kk + khalf*16];
#pragma unroll
    for (int p=0;p<16;p++) bf[p] = wr[p];
    c = __builtin_amdgcn_wmma_f32_16x16x32_f16(false, a, false, bf, (short)0, c, false, false);
  }
#pragma unroll
  for (int j=0;j<8;j++){
    int m2 = tileM*16 + 8*khalf + j;
    out[(size_t)m2*256 + col] = c[j];
  }
}

// generic: C = (A1 [+A2]) @ W^T [+bias]; A row-major MxK (K%32==0, K<=128), W NxK.
__global__ void __launch_bounds__(128) k_gemm_wmma(
    const float* __restrict__ A1, const float* __restrict__ A2,
    const float* __restrict__ W, const float* __restrict__ bias,
    float* __restrict__ Cout, int M, int N, int K){
  __shared__ _Float16 Ws[16*136];    // 16 cols x K<=128, stride 136 halfs
  int tid = threadIdx.x;
  int colBase = blockIdx.y*16;
  for (int i = tid; i < 16*K; i += 128){
    int nl = i / K, k = i - nl*K;
    int n = colBase + nl;
    Ws[nl*136 + k] = (_Float16)((n < N) ? W[(size_t)n*K + k] : 0.f);
  }
  __syncthreads();
  int wave = tid >> 5, lane = tid & 31, khalf = lane >> 4;
  int tileM = blockIdx.x*4 + wave;   // M multiple of 64 in all uses
  int row = tileM*16 + (lane & 15);
  int col = colBase + (lane & 15);
  const float* arow  = A1 + (size_t)row*K;
  const float* arow2 = A2 ? A2 + (size_t)row*K : nullptr;
  v8f c = {};
  for (int kk=0; kk<K; kk+=32){
    if (kk + 32 < K) __builtin_prefetch(arow + kk + 32, 0, 1);
    v16h a, bf;
#pragma unroll
    for (int q=0;q<2;q++){
      int kb = kk + khalf*16 + q*8;
      f4v v0 = *(const f4v*)(arow + kb);
      f4v v1 = *(const f4v*)(arow + kb + 4);
      if (arow2){ v0 += *(const f4v*)(arow2 + kb); v1 += *(const f4v*)(arow2 + kb + 4); }
#pragma unroll
      for (int p=0;p<4;p++){ a[q*8+p] = (_Float16)v0[p]; a[q*8+4+p] = (_Float16)v1[p]; }
    }
    const _Float16* wr = &Ws[(lane&15)*136 + kk + khalf*16];
#pragma unroll
    for (int p=0;p<16;p++) bf[p] = wr[p];
    c = __builtin_amdgcn_wmma_f32_16x16x32_f16(false, a, false, bf, (short)0, c, false, false);
  }
  if (col < N){
    float bv = bias ? bias[col] : 0.f;
#pragma unroll
    for (int j=0;j<8;j++){
      int m2 = tileM*16 + 8*khalf + j;
      Cout[(size_t)m2*N + col] = c[j] + bv;
    }
  }
}

// implicit-GEMM 3x3 conv 64->64 (pad=dil). Weights (32-col slice) live in LDS as f16;
// im2col A tile (64 pixels x 32 k) staged per K-step in LDS (async global->LDS when
// the toolchain exposes it). Each wave computes one 16-row M-tile x two 16-col N-tiles.
// mode 0: relu store; mode 1: out = gate*val; mode 2: out += gate*val.
__global__ void __launch_bounds__(128) k_conv_wmma(
    const float* __restrict__ in, const float* __restrict__ Wgt,
    const float* __restrict__ bn_g, const float* __restrict__ bn_b,
    const float* __restrict__ gate, int gch, int dil, int mode,
    float* __restrict__ out){
  __shared__ _Float16 Ws[32*584];    // 32 cols x K=576, stride 584 halfs (37376 B)
  __shared__ float    As[64*36];     // 64 pixels x 32 k, stride 36 floats  (9216 B)
  int tid = threadIdx.x;
  int colBase = blockIdx.y*32;
  // stage this block's 32 output-channel weight rows, f32 -> f16
  for (int i = tid; i < 32*576; i += 128){
    int nl = i / 576, k = i - nl*576;
    Ws[nl*584 + k] = (_Float16)Wgt[(size_t)(colBase+nl)*576 + k];
  }
  int wave = tid >> 5, lane = tid & 31, khalf = lane >> 4;
  int tileM = blockIdx.x*4 + wave;   // M = 100352 -> exact
  int rowloc = wave*16 + (lane & 15);
  int pixBase = blockIdx.x*64;
  v8f acc[2]; acc[0] = (v8f){}; acc[1] = (v8f){};
  __syncthreads();
  for (int kk=0; kk<576; kk+=32){
    __syncthreads();               // previous tile fully consumed
    // cooperative im2col staging: 2048 elements, coalesced along pixels
    for (int i = tid; i < 64*32; i += 128){
      int pl = i & 63, kl = i >> 6;
      int k = kk + kl;
      int ci = k/9; int t = k - ci*9; int kh = t/3; int kw = t - kh*3;
      int grow = pixBase + pl;
      int gb = grow / HWF, gp = grow - gb*HWF;
      int oh = gp / WWD, ow = gp - oh*WWD;
      int ih = oh + (kh-1)*dil, iw2 = ow + (kw-1)*dil;
      float* lp = &As[pl*36 + kl];
      if (ih>=0 && ih<HH && iw2>=0 && iw2<WWD){
        const float* gp2 = &in[((size_t)(gb*CC+ci)*HH + ih)*WWD + iw2];
#ifdef HAVE_ASYNC_LDS
        __builtin_amdgcn_global_load_async_to_lds_b32(
            (__attribute__((address_space(1))) int*)gp2,
            (__attribute__((address_space(3))) int*)lp, 0, 0);
#else
        *lp = *gp2;
#endif
      } else {
        *lp = 0.f;
      }
    }
#ifdef HAVE_ASYNC_LDS
    __builtin_amdgcn_s_wait_asynccnt(0);
#endif
    __syncthreads();
    // A fragment: two contiguous 8-float runs from LDS, convert to f16
    v16h a;
    const float* ar = &As[rowloc*36];
#pragma unroll
    for (int p=0;p<8;p++){
      a[p]   = (_Float16)ar[p + khalf*16];
      a[p+8] = (_Float16)ar[8 + p + khalf*16];
    }
    // two N-tiles share the A fragment; B fragments are contiguous LDS runs
#pragma unroll
    for (int nt=0; nt<2; nt++){
      const _Float16* wr = &Ws[(nt*16 + (lane&15))*584 + kk + khalf*16];
      v16h bf;
#pragma unroll
      for (int p=0;p<16;p++) bf[p] = wr[p];
      acc[nt] = __builtin_amdgcn_wmma_f32_16x16x32_f16(false, a, false, bf, (short)0, acc[nt], false, false);
    }
  }
#pragma unroll
  for (int j=0;j<8;j++){
    int m2 = tileM*16 + 8*khalf + j;
    int b2 = m2 / HWF, pix2 = m2 - b2*HWF;
    float gvv = (mode != 0) ? gate[(size_t)(b2*3+gch)*HWF + pix2] : 0.f;
#pragma unroll
    for (int nt=0; nt<2; nt++){
      int col = colBase + nt*16 + (lane & 15);
      float v = acc[nt][j]*bn_g[col] + bn_b[col];
      size_t oidx = (size_t)(b2*CC+col)*HWF + pix2;
      if (mode == 0)      out[oidx] = v > 0.f ? v : 0.f;
      else if (mode == 1) out[oidx] = gvv*v;
      else                out[oidx] += gvv*v;
    }
  }
}

// ---------------- mamba kernels ----------------

__global__ void k_cconv(const float* __restrict__ xz, const float* __restrict__ cw,
                        const float* __restrict__ cb, int rev, float* __restrict__ u){
  int idx = blockIdx.x*blockDim.x + threadIdx.x;
  if (idx >= BB*LSEQ*DIN) return;
  int ch = idx & 127, m = idx >> 7;
  int b = m / LSEQ, l = m - b*LSEQ;
  float acc = cb[ch];
#pragma unroll
  for (int j=0;j<4;j++){
    int ls = l - 3 + j;
    if (ls >= 0){
      int pos = rev ? (LSEQ-1-ls) : ls;
      acc += xz[(size_t)(b*LSEQ+pos)*256 + ch] * cw[ch*4+j];
    }
  }
  u[(size_t)idx] = acc * sigmoidf_(acc);
}

__global__ void k_dt(const float* __restrict__ dbl, const float* __restrict__ dtw,
                     const float* __restrict__ dtb, float* __restrict__ dtf){
  int idx = blockIdx.x*blockDim.x + threadIdx.x;
  if (idx >= BB*LSEQ*DIN) return;
  int ch = idx & 127, m = idx >> 7;
  float acc = dtb[ch];
#pragma unroll
  for (int r=0;r<4;r++) acc += dbl[(size_t)m*36 + r]*dtw[ch*4+r];
  dtf[idx] = acc > 20.f ? acc : log1pf(expf(acc));
}

// selective scan: one block per batch, 128 threads = channels, 16 states in regs
__global__ void k_scan(const float* __restrict__ xz, const float* __restrict__ u,
                       const float* __restrict__ dtf, const float* __restrict__ dbl,
                       const float* __restrict__ Alog, const float* __restrict__ Dp,
                       int rev, float* __restrict__ yout){
  int b = blockIdx.x, ch = threadIdx.x;
  float Av[16], h[16];
#pragma unroll
  for (int n=0;n<16;n++){ Av[n] = -expf(Alog[ch*16+n]); h[n] = 0.f; }
  float Dv = Dp[ch];
  __shared__ float sBC[32];
  for (int l=0;l<LSEQ;l++){
    if (ch < 32) sBC[ch] = dbl[(size_t)(b*LSEQ+l)*36 + 4 + ch];
    __syncthreads();
    size_t mi = (size_t)(b*LSEQ+l)*128 + ch;
    float dtv = dtf[mi], uv = u[mi];
    float y = 0.f;
#pragma unroll
    for (int n=0;n<16;n++){
      float dA = expf(dtv*Av[n]);
      h[n] = dA*h[n] + dtv*sBC[n]*uv;
      y += h[n]*sBC[16+n];
    }
    y += uv*Dv;
    int pos = rev ? (LSEQ-1-l) : l;
    float z = xz[(size_t)(b*LSEQ+pos)*256 + 128 + ch];
    yout[(size_t)(b*LSEQ+pos)*128 + ch] = y * (z*sigmoidf_(z));
    __syncthreads();
  }
}

__global__ void k_lnorm(const float* __restrict__ seq, const float* __restrict__ g,
                        const float* __restrict__ bt, float* __restrict__ fm){
  int m = blockIdx.x*blockDim.x + threadIdx.x;
  if (m >= BB*LSEQ) return;
  int b = m / LSEQ, l = m - b*LSEQ;
  float mean = 0.f;
  for (int c=0;c<CC;c++) mean += seq[(size_t)m*CC + c];
  mean *= (1.f/CC);
  float var = 0.f;
  for (int c=0;c<CC;c++){ float d = seq[(size_t)m*CC + c] - mean; var += d*d; }
  float rs = rsqrtf(var*(1.f/CC) + 1e-5f);
  for (int c=0;c<CC;c++)
    fm[(size_t)(b*CC+c)*LSEQ + l] = (seq[(size_t)m*CC + c] - mean)*rs*g[c] + bt[c];
}

__global__ void k_fillum(const float* __restrict__ fm, const float* __restrict__ res,
                         float* __restrict__ out1){
  int idx = blockIdx.x*blockDim.x + threadIdx.x;
  if (idx >= BB*CC*HWF) return;
  int bc = idx / HWF, pix = idx - bc*HWF;
  int h = pix / WWD, w = pix - h*WWD;
  out1[idx] = bil2x(fm + (size_t)bc*HWH, HP, WP, h, w) + res[idx];
}

// ---------------- gating / AFF ----------------

__global__ void k_mean_hw(const float* __restrict__ in1, const float* __restrict__ in2,
                          float* __restrict__ outv, int n){
  int bc = blockIdx.x, t = threadIdx.x;
  const float* p1 = in1 + (size_t)bc*n;
  const float* p2 = in2 ? in2 + (size_t)bc*n : nullptr;
  float s = 0.f;
  for (int i=t;i<n;i+=blockDim.x){ float v = p1[i]; if (p2) v += p2[i]; s += v; }
  __shared__ float red[256];
  red[t] = s; __syncthreads();
  for (int off=128; off>0; off>>=1){ if (t<off) red[t]+=red[t+off]; __syncthreads(); }
  if (t==0) outv[bc] = red[0]/(float)n;
}

__global__ void k_small_mlp(const float* __restrict__ invec, const float* __restrict__ w1,
                            const float* __restrict__ b1, const float* __restrict__ w2,
                            const float* __restrict__ b2, float* __restrict__ outv,
                            int inC, int hidC, int outC, float slope){
  int b = blockIdx.x, t = threadIdx.x;
  __shared__ float hid[32];
  if (t < hidC){
    float a = b1[t];
    for (int c=0;c<inC;c++) a += w1[t*inC+c]*invec[b*inC+c];
    hid[t] = a > 0.f ? a : slope*a;
  }
  __syncthreads();
  if (t < outC){
    float a = b2[t];
    for (int r=0;r<hidC;r++) a += w2[t*hidC+r]*hid[r];
    outv[b*outC+t] = a;
  }
}

__global__ void k_gate(const float* __restrict__ fh, const float* __restrict__ grad,
                       const float* __restrict__ pw, const float* __restrict__ glv,
                       float* __restrict__ gate){
  int idx = blockIdx.x*blockDim.x + threadIdx.x;
  if (idx >= BB*HWF) return;
  int b = idx / HWF, pix = idx - b*HWF;
  int h = pix / WWD, w = pix - h*WWD;
  float g = bil2x(grad + (size_t)b*HWH, HP, WP, h, w);
  float s0 = pw[64]*g     + glv[b*3+0];
  float s1 = pw[65+64]*g  + glv[b*3+1];
  float s2 = pw[130+64]*g + glv[b*3+2];
  const float* fb = fh + (size_t)b*CC*HWF + pix;
  for (int c=0;c<CC;c++){
    float xv = fb[(size_t)c*HWF];
    s0 += pw[c]*xv; s1 += pw[65+c]*xv; s2 += pw[130+c]*xv;
  }
  float mx = fmaxf(s0, fmaxf(s1, s2));
  float e0 = expf(s0-mx), e1 = expf(s1-mx), e2 = expf(s2-mx);
  float inv = 1.f/(e0+e1+e2);
  gate[(size_t)(b*3+0)*HWF + pix] = e0*inv;
  gate[(size_t)(b*3+1)*HWF + pix] = e1*inv;
  gate[(size_t)(b*3+2)*HWF + pix] = e2*inv;
}

__global__ void k_lap_final(const float* __restrict__ fh, const float* __restrict__ lap,
                            const float* __restrict__ gate, const float* __restrict__ acc,
                            float* __restrict__ out2){
  int idx = blockIdx.x*blockDim.x + threadIdx.x;
  if (idx >= BB*CC*HWF) return;
  int bc = idx / HWF, pix = idx - bc*HWF;
  int b = bc / CC, c = bc - b*CC;
  int h = pix / WWD, w = pix - h*WWD;
  const float* p = fh + (size_t)bc*HWF;
  float e2 = 0.f;
  for (int kh=0;kh<3;kh++){
    int ih = h + kh - 1; if (ih < 0 || ih >= HH) continue;
    for (int kw=0;kw<3;kw++){
      int iw2 = w + kw - 1; if (iw2 < 0 || iw2 >= WWD) continue;
      e2 += p[ih*WWD + iw2] * lap[c*9 + kh*3 + kw];
    }
  }
  float v = acc[idx] + gate[(size_t)(b*3+2)*HWF + pix]*e2;
  out2[idx] = v > 0.f ? v : 0.f;
}

__global__ void k_aff(const float* __restrict__ fi, const float* __restrict__ fg,
                      const float* __restrict__ l1w, const float* __restrict__ l1b,
                      const float* __restrict__ l2w, const float* __restrict__ l2b,
                      const float* __restrict__ xg, float* __restrict__ outp){
  int idx = blockIdx.x*blockDim.x + threadIdx.x;
  if (idx >= BB*HWF) return;
  int b = idx / HWF, pix = idx - b*HWF;
  const float* fib = fi + (size_t)b*CC*HWF + pix;
  const float* fgb = fg + (size_t)b*CC*HWF + pix;
  float hs[16];
#pragma unroll
  for (int r=0;r<16;r++) hs[r] = l1b[r];
  for (int c=0;c<CC;c++){
    float xa = fib[(size_t)c*HWF] + fgb[(size_t)c*HWF];
#pragma unroll
    for (int r=0;r<16;r++) hs[r] += l1w[r*64+c]*xa;
  }
#pragma unroll
  for (int r=0;r<16;r++) hs[r] = hs[r] > 0.f ? hs[r] : 0.1f*hs[r];
  for (int c=0;c<CC;c++){
    float xl = l2b[c];
#pragma unroll
    for (int r=0;r<16;r++) xl += l2w[c*16+r]*hs[r];
    float wei = sigmoidf_(xl + xg[b*CC+c]);
    float a = fib[(size_t)c*HWF], g2 = fgb[(size_t)c*HWF];
    outp[(size_t)(b*CC+c)*HWF + pix] = 2.f*a*wei + 2.f*g2*(1.f-wei);
  }
}

// ---------------- launch ----------------

extern "C" void kernel_launch(void* const* d_in, const int* in_sizes, int n_in,
                              void* d_out, int out_size, void* d_ws, size_t ws_size,
                              hipStream_t stream) {
  const float* x        = (const float*)d_in[0];
  const float* illum    = (const float*)d_in[1];
  const float* grad     = (const float*)d_in[2];
  const float* illum_w  = (const float*)d_in[3];
  const float* illum_b  = (const float*)d_in[4];
  const float* dw_w     = (const float*)d_in[5];
  const float* dw_b     = (const float*)d_in[6];
  const float* rms_g    = (const float*)d_in[7];
  const float* in_w     = (const float*)d_in[8];
  const float* f_conv_w = (const float*)d_in[9];
  const float* f_conv_b = (const float*)d_in[10];
  const float* f_xproj_w= (const float*)d_in[11];
  const float* f_dt_w   = (const float*)d_in[12];
  const float* f_dt_b   = (const float*)d_in[13];
  const float* f_A_log  = (const float*)d_in[14];
  const float* f_D      = (const float*)d_in[15];
  const float* b_conv_w = (const float*)d_in[16];
  const float* b_conv_b = (const float*)d_in[17];
  const float* b_xproj_w= (const float*)d_in[18];
  const float* b_dt_w   = (const float*)d_in[19];
  const float* b_dt_b   = (const float*)d_in[20];
  const float* b_A_log  = (const float*)d_in[21];
  const float* b_D      = (const float*)d_in[22];
  const float* out_w    = (const float*)d_in[23];
  const float* ln_g     = (const float*)d_in[24];
  const float* ln_b     = (const float*)d_in[25];
  const float* alpha_p  = (const float*)d_in[26];
  const float* e0_w1    = (const float*)d_in[27];
  const float* e0_w2    = (const float*)d_in[28];
  const float* e0_bn1_g = (const float*)d_in[29];
  const float* e0_bn1_b = (const float*)d_in[30];
  const float* e0_bn2_g = (const float*)d_in[31];
  const float* e0_bn2_b = (const float*)d_in[32];
  const float* e1_w1    = (const float*)d_in[33];
  const float* e1_w2    = (const float*)d_in[34];
  const float* e1_bn1_g = (const float*)d_in[35];
  const float* e1_bn1_b = (const float*)d_in[36];
  const float* e1_bn2_g = (const float*)d_in[37];
  const float* e1_bn2_b = (const float*)d_in[38];
  const float* lap_w    = (const float*)d_in[39];
  const float* proj_w   = (const float*)d_in[40];
  const float* gp1_w    = (const float*)d_in[41];
  const float* gp1_b    = (const float*)d_in[42];
  const float* gp2_w    = (const float*)d_in[43];
  const float* gp2_b    = (const float*)d_in[44];
  const float* aff_l1_w = (const float*)d_in[45];
  const float* aff_l1_b = (const float*)d_in[46];
  const float* aff_l2_w = (const float*)d_in[47];
  const float* aff_l2_b = (const float*)d_in[48];
  const float* aff_g1_w = (const float*)d_in[49];
  const float* aff_g1_b = (const float*)d_in[50];
  const float* aff_g2_w = (const float*)d_in[51];
  const float* aff_g2_b = (const float*)d_in[52];

  const size_t nFull = (size_t)BB*CC*HWF;   // 6,422,528
  const size_t nHalf = (size_t)BB*CC*HWH;
  const int    M1    = BB*LSEQ;             // 25,088

  float* ws = (float*)d_ws;
  size_t o = 0;
  float* f_low  = ws + o; o += nFull;
  float* f_high = ws + o; o += nFull;
  float* fd     = ws + o; o += nHalf;
  float* res    = ws + o; o += nFull;
  float* tmp    = ws + o; o += nFull;
  float* xz     = ws + o; o += (size_t)M1*256;
  float* ubuf   = ws + o; o += (size_t)M1*128;
  float* dbl    = ws + o; o += (size_t)M1*36;
  float* dtf    = ws + o; o += (size_t)M1*128;
  float* yf     = ws + o; o += (size_t)M1*128;
  float* yb     = ws + o; o += (size_t)M1*128;
  float* seq    = ws + o; o += (size_t)M1*64;
  float* fm     = ws + o; o += nHalf;
  float* rmss   = ws + o; o += (size_t)M1;
  float* gate   = ws + o; o += (size_t)BB*3*HWF;
  float* gvec   = ws + o; o += 128;
  float* glv    = ws + o; o += 8;
  float* ga     = ws + o; o += 128;
  float* xg     = ws + o; o += 128;

  float* outp    = (float*)d_out;
  float* f_out   = outp;
  float* f_illum = outp + nFull;
  float* f_grad  = outp + 2*nFull;

  const int T = 256;
  // stage 1: illumination gating + high branch
  k_pre<<<CDIV(BB*HWF, T), T, 0, stream>>>(x, illum, grad, illum_w, illum_b, alpha_p, f_low, f_high);
  // stage 2: avgpool / residual / depthwise
  k_avgpool<<<CDIV((int)nHalf, T), T, 0, stream>>>(f_low, fd);
  k_resup  <<<CDIV((int)nFull, T), T, 0, stream>>>(f_low, fd, tmp);
  k_dw3    <<<CDIV((int)nFull, T), T, 0, stream>>>(tmp, dw_w, dw_b, res);
  // stage 3: mamba
  k_rmsscale<<<CDIV(M1, T), T, 0, stream>>>(fd, rmss);
  {
    dim3 g(M1/64, 16);
    k_gemm_inproj<<<g, 128, 0, stream>>>(fd, rmss, rms_g, in_w, xz);
  }
  // forward branch
  k_cconv<<<CDIV(M1*128, T), T, 0, stream>>>(xz, f_conv_w, f_conv_b, 0, ubuf);
  {
    dim3 g(M1/64, 3);
    k_gemm_wmma<<<g, 128, 0, stream>>>(ubuf, nullptr, f_xproj_w, nullptr, dbl, M1, 36, 128);
  }
  k_dt<<<CDIV(M1*128, T), T, 0, stream>>>(dbl, f_dt_w, f_dt_b, dtf);
  k_scan<<<BB, 128, 0, stream>>>(xz, ubuf, dtf, dbl, f_A_log, f_D, 0, yf);
  // backward branch (reuses ubuf/dbl/dtf)
  k_cconv<<<CDIV(M1*128, T), T, 0, stream>>>(xz, b_conv_w, b_conv_b, 1, ubuf);
  {
    dim3 g(M1/64, 3);
    k_gemm_wmma<<<g, 128, 0, stream>>>(ubuf, nullptr, b_xproj_w, nullptr, dbl, M1, 36, 128);
  }
  k_dt<<<CDIV(M1*128, T), T, 0, stream>>>(dbl, b_dt_w, b_dt_b, dtf);
  k_scan<<<BB, 128, 0, stream>>>(xz, ubuf, dtf, dbl, b_A_log, b_D, 1, yb);
  // out proj + layernorm + upsample
  {
    dim3 g(M1/64, 4);
    k_gemm_wmma<<<g, 128, 0, stream>>>(yf, yb, out_w, nullptr, seq, M1, 64, 128);
  }
  k_lnorm <<<CDIV(M1, T), T, 0, stream>>>(seq, ln_g, ln_b, fm);
  k_fillum<<<CDIV((int)nFull, T), T, 0, stream>>>(fm, res, f_illum);
  // stage 4: expert gating
  k_mean_hw  <<<BB*CC, 256, 0, stream>>>(f_high, nullptr, gvec, HWF);
  k_small_mlp<<<BB, 64, 0, stream>>>(gvec, gp1_w, gp1_b, gp2_w, gp2_b, glv, 64, 16, 3, 0.f);
  k_gate     <<<CDIV(BB*HWF, T), T, 0, stream>>>(f_high, grad, proj_w, glv, gate);
  // experts (WMMA implicit-GEMM convs; LDS-resident weights + staged im2col tiles)
  {
    dim3 g(BB*HWF/64, 2);
    k_conv_wmma<<<g, 128, 0, stream>>>(f_high, e0_w1, e0_bn1_g, e0_bn1_b, nullptr, 0, 1, 0, tmp);
    k_conv_wmma<<<g, 128, 0, stream>>>(tmp,    e0_w2, e0_bn2_g, e0_bn2_b, gate,    0, 1, 1, f_grad);
    k_conv_wmma<<<g, 128, 0, stream>>>(f_high, e1_w1, e1_bn1_g, e1_bn1_b, nullptr, 0, 2, 0, tmp);
    k_conv_wmma<<<g, 128, 0, stream>>>(tmp,    e1_w2, e1_bn2_g, e1_bn2_b, gate,    1, 2, 2, f_grad);
  }
  k_lap_final<<<CDIV((int)nFull, T), T, 0, stream>>>(f_high, lap_w, gate, f_grad, f_grad);
  // stage 5: AFF fusion
  k_mean_hw  <<<BB*CC, 256, 0, stream>>>(f_illum, f_grad, ga, HWF);
  k_small_mlp<<<BB, 64, 0, stream>>>(ga, aff_g1_w, aff_g1_b, aff_g2_w, aff_g2_b, xg, 64, 16, 64, 0.1f);
  k_aff      <<<CDIV(BB*HWF, T), T, 0, stream>>>(f_illum, f_grad, aff_l1_w, aff_l1_b,
                                                 aff_l2_w, aff_l2_b, xg, f_out);
}
